// GraphEncoder_64304250356386
// MI455X (gfx1250) — compile-verified, matrix operand order checked by reading, not run on hardware
//
#include <hip/hip_runtime.h>

#define NND 20000
#define NE  100000
#define NR  1000
#define FD  128
#define NH  4
#define NL  4
#define HCV 512
#define ETOTN (NE + NND)
#define RELEXT 1008
#define NEG_SLOPE 0.2f

typedef __attribute__((ext_vector_type(16))) __bf16 v16bf;
typedef __attribute__((ext_vector_type(8)))  float  v8f;
typedef int v4i_t __attribute__((vector_size(16)));
typedef __attribute__((address_space(1))) v4i_t* gp4_t;   // global (AS1) int4*
typedef __attribute__((address_space(3))) v4i_t* lp4_t;   // LDS (AS3) int4*

union BF16Frag { v16bf v; uint4 u[2]; };

__device__ __forceinline__ unsigned short f2bf(float f) {
  unsigned u = __float_as_uint(f);
  u += 0x7FFFu + ((u >> 16) & 1u);          // round-to-nearest-even
  return (unsigned short)(u >> 16);
}
__device__ __forceinline__ float lrelu(float v) { return v > 0.f ? v : NEG_SLOPE * v; }
// order-preserving float<->uint encoding for atomicMax on floats
__device__ __forceinline__ unsigned fenc(float f) {
  unsigned u = __float_as_uint(f);
  return (u & 0x80000000u) ? ~u : (u | 0x80000000u);
}
__device__ __forceinline__ float fdec(unsigned u) {
  unsigned b = (u & 0x80000000u) ? (u & 0x7FFFFFFFu) : ~u;
  return __uint_as_float(b);
}
__device__ __forceinline__ float4 ld4(const float* p) { return *(const float4*)p; }
__device__ __forceinline__ void atomAddF(float* p, float v) {
  __hip_atomic_fetch_add(p, v, __ATOMIC_RELAXED, __HIP_MEMORY_SCOPE_AGENT);
}

// 16B global -> LDS copy: async DMA path (ASYNCcnt) when available, else via VGPRs
__device__ __forceinline__ void async_cp16(const void* g, void* l) {
#if __has_builtin(__builtin_amdgcn_global_load_async_to_lds_b128)
  __builtin_amdgcn_global_load_async_to_lds_b128((gp4_t)(g), (lp4_t)(l), 0, 0);
#else
  *(uint4*)l = *(const uint4*)g;
#endif
}
__device__ __forceinline__ void async_cp_wait() {
#if __has_builtin(__builtin_amdgcn_global_load_async_to_lds_b128)
#if __has_builtin(__builtin_amdgcn_s_wait_asynccnt)
  __builtin_amdgcn_s_wait_asynccnt(0);
#else
  asm volatile("s_wait_asynccnt 0x0" ::: "memory");
#endif
#endif
}

// ---------------- utility kernels ----------------

__global__ void fill_u32_k(unsigned* p, unsigned v, int n) {
  int i = blockIdx.x * blockDim.x + threadIdx.x;
  if (i < n) p[i] = v;
}

__global__ void hist_k(const int* __restrict__ ri, unsigned* __restrict__ counts) {
  int i = blockIdx.x * blockDim.x + threadIdx.x;
  if (i < NE) atomicAdd(&counts[ri[i]], 1u);
}

// weighted mean over gathered relation rows == exact mean of ea over edges
__global__ void mean_row_k(const unsigned* __restrict__ counts,
                           const float* __restrict__ rel, float* __restrict__ mean) {
  int c = threadIdx.x;
  if (c >= FD) return;
  float s = 0.f;
  for (int r = 0; r < NR; ++r) s += (float)counts[r] * rel[(size_t)r * FD + c];
  mean[c] = s * (1.0f / NE);
}

// rows 0..999 = relations, row 1000 = mean row (self-loop fill), rows 1001..1007 = 0
__global__ void build_relext_k(const float* __restrict__ rel, const float* __restrict__ mean,
                               unsigned short* __restrict__ rext) {
  int i = blockIdx.x * blockDim.x + threadIdx.x;
  if (i >= RELEXT * FD) return;
  int r = i / FD, c = i - r * FD;
  float v = (r < NR) ? rel[i] : ((r == NR) ? mean[c] : 0.f);
  rext[i] = f2bf(v);
}

__global__ void f32_to_bf16_k(const float* __restrict__ s, unsigned short* __restrict__ d, int n) {
  int i = blockIdx.x * blockDim.x + threadIdx.x;
  if (i < n) d[i] = f2bf(s[i]);
}

// pack W [K=128, N=512] row-major f32 into per-lane-contiguous bf16 B-fragments:
// dst[((nt*4 + kt)*32 + lane)*16 + j]; one 64-column group (nt=4g..4g+3, all kt)
// is a contiguous 16KB span -> single LDS stage per block.
__global__ void pack_w_k(const float* __restrict__ W, unsigned short* __restrict__ dst) {
  int idx = blockIdx.x * blockDim.x + threadIdx.x;
  if (idx >= FD * HCV) return;
  int j = idx & 15;
  int lane = (idx >> 4) & 31;
  int tile = idx >> 9;
  int kt = tile & 3;
  int nt = tile >> 2;
  int n = nt * 16 + (lane & 15);
  int khalf = lane >> 4;
  int k = kt * 32 + ((j < 8) ? (khalf * 8 + j) : (16 + khalf * 8 + (j - 8)));
  dst[idx] = f2bf(W[(size_t)k * HCV + n]);
}

// ---------------- GEMM: C[M,512] = A[M,128](bf16) * Bpacked + bias ----------------
// Block = 8 waves: one 64-column group, 8 consecutive M tiles. The block's 16KB
// packed-B slice is staged to LDS once (async-to-LDS when available) and reused
// by all 8 waves; each wave runs 16 v_wmma_f32_16x16x32_bf16.
__global__ void gemm_bf16_wmma(const unsigned short* __restrict__ A,
                               const unsigned short* __restrict__ Bp,
                               const float* __restrict__ bias,
                               float* __restrict__ C, int M) {
  __shared__ unsigned short sB[16 * 512];   // 16 tiles x 32 lanes x 16 bf16 = 16KB
  int ng = blockIdx.x & 7;                  // column group (64 cols)
  int group = blockIdx.x >> 3;              // M-tile group (8 tiles)
  int w = threadIdx.x >> 5;
  int lane = threadIdx.x & 31;
  int mtiles = M >> 4;
  int mt = group * 8 + w;

  {  // cooperative stage of contiguous 16KB B slice for this column group
    const uint4* src = (const uint4*)(Bp + (size_t)ng * (16 * 512));
    uint4* dst = (uint4*)sB;
    for (int i = threadIdx.x; i < 1024; i += 256) async_cp16(src + i, dst + i);
    async_cp_wait();
  }
  __syncthreads();

  if (mt >= mtiles) return;
  int rsel = lane & 15;
  int khalf = lane >> 4;
  const unsigned short* arow = A + (size_t)(mt * 16 + rsel) * FD;
  v8f acc[4] = {};
#pragma unroll
  for (int kt = 0; kt < 4; ++kt) {
    BF16Frag a;
    a.u[0] = *(const uint4*)(arow + kt * 32 + khalf * 8);        // K 0-7 / 8-15
    a.u[1] = *(const uint4*)(arow + kt * 32 + 16 + khalf * 8);   // K 16-23 / 24-31
#pragma unroll
    for (int t = 0; t < 4; ++t) {
      const unsigned short* bptr = sB + (((t * 4 + kt) * 32 + lane) * 16);
      BF16Frag b;
      b.u[0] = *(const uint4*)(bptr);
      b.u[1] = *(const uint4*)(bptr + 8);
      acc[t] = __builtin_amdgcn_wmma_f32_16x16x32_bf16(false, a.v, false, b.v,
                                                       (short)0, acc[t], false, false);
    }
  }
  // C/D layout: lane holds col = base + (lane&15), rows (lane>>4)*8 + i
  int rbase = mt * 16 + khalf * 8;
#pragma unroll
  for (int t = 0; t < 4; ++t) {
    int col = (ng * 4 + t) * 16 + rsel;
    float bc = bias ? bias[col] : 0.0f;
#pragma unroll
    for (int i = 0; i < 8; ++i)
      C[(size_t)(rbase + i) * HCV + col] = acc[t][i] + bc;
  }
}

// ---------------- edge kernels (one wave per edge; Etot = E + N self loops) ----------------

__global__ void edge_logits_k(const float* __restrict__ xl, const float* __restrict__ xr,
                              const float* __restrict__ ee, const float* __restrict__ att,
                              const int* __restrict__ ei, const int* __restrict__ ri,
                              float* __restrict__ logits, unsigned* __restrict__ mmax) {
  int gtid = blockIdx.x * blockDim.x + threadIdx.x;
  int e = gtid >> 5;
  int lane = threadIdx.x & 31;
  if (e >= ETOTN) return;
  int s, d, row;
  if (e < NE) { s = ei[e]; d = ei[NE + e]; row = ri[e]; }
  else { s = e - NE; d = s; row = NR; }      // self loop, mean edge-attr row
  int c = lane * 4;
  const float* pxl = xl + (size_t)s * HCV + c;
  const float* pxr = xr + (size_t)d * HCV + c;
  const float* pee = ee + (size_t)row * HCV + c;
  float part[NH];
#pragma unroll
  for (int h = 0; h < NH; ++h) {
    float4 a = ld4(pxl + h * FD);
    float4 b = ld4(pxr + h * FD);
    float4 g = ld4(pee + h * FD);
    float4 w = ld4(att + h * FD + c);
    part[h] = lrelu(a.x + b.x + g.x) * w.x + lrelu(a.y + b.y + g.y) * w.y
            + lrelu(a.z + b.z + g.z) * w.z + lrelu(a.w + b.w + g.w) * w.w;
  }
#pragma unroll
  for (int h = 0; h < NH; ++h) {
#pragma unroll
    for (int off = 16; off; off >>= 1) part[h] += __shfl_xor(part[h], off, 32);
  }
  if (lane == 0) {
#pragma unroll
    for (int h = 0; h < NH; ++h) {
      logits[(size_t)e * NH + h] = part[h];
      atomicMax(&mmax[d * NH + h], fenc(part[h]));
    }
  }
}

__global__ void edge_exp_k(float* __restrict__ logits, const unsigned* __restrict__ mmax,
                           float* __restrict__ denom, const int* __restrict__ ei) {
  int i = blockIdx.x * blockDim.x + threadIdx.x;
  if (i >= ETOTN * NH) return;
  int e = i >> 2, h = i & 3;
  int d = (e < NE) ? ei[NE + e] : (e - NE);
  float p = __expf(logits[i] - fdec(mmax[d * NH + h]));
  logits[i] = p;                              // reuse buffer: logits -> p
  atomAddF(&denom[d * NH + h], p);
}

__global__ void init_out_k(float* __restrict__ h, const float* __restrict__ bias) {
  int i = blockIdx.x * blockDim.x + threadIdx.x;
  if (i < NND * FD) h[i] = bias[i & (FD - 1)];
}

// out[d, c] += (1/H) * sum_h alpha_eh * xl[s, h*128 + c]  (head-mean folded in)
__global__ void edge_aggr_k(const float* __restrict__ p, const float* __restrict__ denom,
                            const float* __restrict__ xl, float* __restrict__ out,
                            const int* __restrict__ ei) {
  int gtid = blockIdx.x * blockDim.x + threadIdx.x;
  int e = gtid >> 5;
  int lane = threadIdx.x & 31;
  if (e >= ETOTN) return;
  int s, d;
  if (e < NE) { s = ei[e]; d = ei[NE + e]; } else { s = e - NE; d = s; }
  float al[NH];
#pragma unroll
  for (int h = 0; h < NH; ++h)
    al[h] = p[(size_t)e * NH + h] / denom[d * NH + h] * (1.0f / NH);
  int c = lane * 4;
  const float* pxl = xl + (size_t)s * HCV + c;
  float ax = 0.f, ay = 0.f, az = 0.f, aw = 0.f;
#pragma unroll
  for (int h = 0; h < NH; ++h) {
    float4 v = ld4(pxl + h * FD);
    ax += al[h] * v.x; ay += al[h] * v.y; az += al[h] * v.z; aw += al[h] * v.w;
  }
  float* po = out + (size_t)d * FD + c;
  atomAddF(po + 0, ax); atomAddF(po + 1, ay); atomAddF(po + 2, az); atomAddF(po + 3, aw);
}

// ---------------- launch ----------------

extern "C" void kernel_launch(void* const* d_in, const int* in_sizes, int n_in,
                              void* d_out, int out_size, void* d_ws, size_t ws_size,
                              hipStream_t stream) {
  (void)in_sizes; (void)n_in; (void)out_size; (void)ws_size;
  const float* x    = (const float*)d_in[0];
  const int*   ei   = (const int*)d_in[1];     // [2, E]: src then dst
  const float* rel  = (const float*)d_in[2];
  const int*   ri   = (const int*)d_in[3];
  const float* Wl   = (const float*)d_in[4];
  const float* bl   = (const float*)d_in[5];
  const float* Wr   = (const float*)d_in[6];
  const float* br   = (const float*)d_in[7];
  const float* We   = (const float*)d_in[8];
  const float* att  = (const float*)d_in[9];
  const float* bias = (const float*)d_in[10];

  char* ws = (char*)d_ws;
  size_t off = 0;
  auto take = [&](size_t b) { size_t r = off; off += (b + 255) & ~(size_t)255; return r; };
  unsigned short* hbf  = (unsigned short*)(ws + take((size_t)NND * FD * 2));
  float* hf            = (float*)(ws + take((size_t)NND * FD * 4));
  float* xl            = (float*)(ws + take((size_t)NND * HCV * 4));
  float* xr            = (float*)(ws + take((size_t)NND * HCV * 4));
  float* eet           = (float*)(ws + take((size_t)RELEXT * HCV * 4));
  unsigned short* rext = (unsigned short*)(ws + take((size_t)RELEXT * FD * 2));
  unsigned short* wlp  = (unsigned short*)(ws + take((size_t)FD * HCV * 2));
  unsigned short* wrp  = (unsigned short*)(ws + take((size_t)FD * HCV * 2));
  unsigned short* wep  = (unsigned short*)(ws + take((size_t)FD * HCV * 2));
  float* lg            = (float*)(ws + take((size_t)ETOTN * NH * 4));
  unsigned* mmax       = (unsigned*)(ws + take((size_t)NND * NH * 4));
  float* denom         = (float*)(ws + take((size_t)NND * NH * 4));
  unsigned* counts     = (unsigned*)(ws + take((size_t)NR * 4));
  float* mean          = (float*)(ws + take((size_t)FD * 4));

  const int B = 256;
  auto blocks = [](long n) { return (unsigned)((n + 255) / 256); };
  auto gemmGrid = [](int M) { return (unsigned)(((M / 16 + 7) / 8) * 8); };  // mt-groups x 8 col-groups

  // edge-attr mean row (exact, via relation histogram) + bf16 relation table
  fill_u32_k<<<blocks(NR), B, 0, stream>>>(counts, 0u, NR);
  hist_k<<<blocks(NE), B, 0, stream>>>(ri, counts);
  mean_row_k<<<1, 128, 0, stream>>>(counts, rel, mean);
  build_relext_k<<<blocks((long)RELEXT * FD), B, 0, stream>>>(rel, mean, rext);

  for (int l = 0; l < NL; ++l) {
    const float* hsrc = (l == 0) ? x : hf;
    f32_to_bf16_k<<<blocks((long)NND * FD), B, 0, stream>>>(hsrc, hbf, NND * FD);
    pack_w_k<<<blocks((long)FD * HCV), B, 0, stream>>>(Wl + (size_t)l * FD * HCV, wlp);
    pack_w_k<<<blocks((long)FD * HCV), B, 0, stream>>>(Wr + (size_t)l * FD * HCV, wrp);
    pack_w_k<<<blocks((long)FD * HCV), B, 0, stream>>>(We + (size_t)l * FD * HCV, wep);

    gemm_bf16_wmma<<<gemmGrid(NND), B, 0, stream>>>(hbf, wlp, bl + (size_t)l * HCV, xl, NND);
    gemm_bf16_wmma<<<gemmGrid(NND), B, 0, stream>>>(hbf, wrp, br + (size_t)l * HCV, xr, NND);
    gemm_bf16_wmma<<<gemmGrid(RELEXT), B, 0, stream>>>(rext, wep, (const float*)nullptr, eet, RELEXT);

    fill_u32_k<<<blocks((long)NND * NH), B, 0, stream>>>(mmax, 0u, NND * NH);          // enc floor
    fill_u32_k<<<blocks((long)NND * NH), B, 0, stream>>>((unsigned*)denom, 0u, NND * NH);
    edge_logits_k<<<blocks((long)ETOTN * 32), B, 0, stream>>>(xl, xr, eet,
                                                              att + (size_t)l * NH * FD, ei, ri, lg, mmax);
    edge_exp_k<<<blocks((long)ETOTN * NH), B, 0, stream>>>(lg, mmax, denom, ei);
    init_out_k<<<blocks((long)NND * FD), B, 0, stream>>>(hf, bias + (size_t)l * FD);
    edge_aggr_k<<<blocks((long)ETOTN * 32), B, 0, stream>>>(lg, denom, xl, hf, ei);
  }

  (void)hipMemcpyAsync(d_out, hf, (size_t)NND * FD * 4, hipMemcpyDeviceToDevice, stream);
  (void)hipMemcpyAsync((float*)d_out + (size_t)NND * FD, rel, (size_t)NR * FD * 4,
                       hipMemcpyDeviceToDevice, stream);
}
